// GraphTransformer2_3504693313909
// MI455X (gfx1250) — compile-verified, matrix operand order checked by reading, not run on hardware
//
#include <hip/hip_runtime.h>
#include <hip/hip_bf16.h>
#include <math.h>

#define N_NODES  16384
#define N_GRAPHS 32
#define SEQ      512
#define HID      128
#define HEADS    4
#define N_EDGESC 262144
#define OUT_CH   64

typedef __bf16 bf16x2  __attribute__((ext_vector_type(2)));
typedef __bf16 bf16x4  __attribute__((ext_vector_type(4)));
typedef __bf16 bf16x8  __attribute__((ext_vector_type(8)));
typedef __bf16 bf16x16 __attribute__((ext_vector_type(16)));
typedef float  f32x4   __attribute__((ext_vector_type(4)));
typedef float  f32x8   __attribute__((ext_vector_type(8)));

__device__ __forceinline__ __bf16 f2bf(float x) {
  union { float f; unsigned u; } a; a.f = x;
  unsigned r = a.u + 0x7fffu + ((a.u >> 16) & 1u);   // round-to-nearest-even
  union { unsigned short s; __bf16 b; } c; c.s = (unsigned short)(r >> 16);
  return c.b;
}

// packed f32x2 -> bf16x2 (v_cvt_pk_bf16_f32 when the target exposes it)
__device__ __forceinline__ bf16x2 pk2(float a, float b) {
#if __has_builtin(__builtin_amdgcn_cvt_pk_bf16_f32)
  auto r = __builtin_amdgcn_cvt_pk_bf16_f32(a, b);
  return __builtin_bit_cast(bf16x2, r);
#else
  bf16x2 o; o[0] = f2bf(a); o[1] = f2bf(b); return o;
#endif
}

// ---- CDNA5 async global->LDS DMA (ASYNCcnt path), GVS addressing -------------
__device__ __forceinline__ void async_copy16(const void* gbase, unsigned voff_bytes,
                                             void* ldsptr) {
  unsigned lds = (unsigned)(uintptr_t)ldsptr;           // addr[31:0] == LDS offset
  asm volatile("global_load_async_to_lds_b128 %0, %1, %2"
               :: "v"(lds), "v"(voff_bytes),
                  "s"((unsigned long long)(uintptr_t)gbase)
               : "memory");
}
__device__ __forceinline__ void wait_async() {
  asm volatile("s_wait_asynccnt 0x0" ::: "memory");
}

// Build a 16x32 (A) / 32x16 (B) bf16 WMMA fragment from a row-major LDS row.
__device__ __forceinline__ bf16x16 frag16(const __bf16* row, int kbase) {
  bf16x8 lo = *(const bf16x8*)(row + kbase);
  bf16x8 hi = *(const bf16x8*)(row + kbase + 16);
  return __builtin_shufflevector(lo, hi, 0,1,2,3,4,5,6,7,8,9,10,11,12,13,14,15);
}

// ---------------------------------------------------------------------------
// f32 -> bf16 tensor conversion (one pass per tensor per call)
// ---------------------------------------------------------------------------
__global__ void cvt_bf16(const float* __restrict__ in, __bf16* __restrict__ out, int n)
{
  int i = (blockIdx.x * blockDim.x + threadIdx.x) * 8;
  if (i >= n) return;
  __builtin_prefetch((const void*)(in + i + 2048), 0, 3);
  f32x4 a = *(const f32x4*)(in + i);
  f32x4 b = *(const f32x4*)(in + i + 4);
  bf16x2 p0 = pk2(a[0], a[1]), p1 = pk2(a[2], a[3]);
  bf16x2 p2 = pk2(b[0], b[1]), p3 = pk2(b[2], b[3]);
  bf16x8 o;
  o[0]=p0[0]; o[1]=p0[1]; o[2]=p1[0]; o[3]=p1[1];
  o[4]=p2[0]; o[5]=p2[1]; o[6]=p3[0]; o[7]=p3[1];
  *(bf16x8*)(out + i) = o;
}

// ---------------------------------------------------------------------------
// Tiled GEMM: C[M,N] = relu?( A[M,K] @ W[N,K]^T + bias ), bf16 inputs, f32 out.
// Block = 256 thr = 8 waves; tile 64x32; one 16x16 WMMA tile per wave.
// Tiles staged with global_load_async_to_lds_b128.
// ---------------------------------------------------------------------------
__global__ __launch_bounds__(256) void gemm_wmma(
    const __bf16* __restrict__ A, const __bf16* __restrict__ W,
    const float* __restrict__ bias, float* __restrict__ C,
    int M, int Nn, int K, int relu)
{
  __shared__ __align__(16) __bf16 As[64][32];
  __shared__ __align__(16) __bf16 Ws[32][32];
  const int tid  = threadIdx.x;
  const int lane = tid & 31, wave = tid >> 5;
  const int wr = wave >> 1, wc = wave & 1;
  const int bm0 = blockIdx.x * 64, bn0 = blockIdx.y * 32;
  const int kbase = (lane < 16) ? 0 : 8;
  f32x8 acc = {};

  for (int k0 = 0; k0 < K; k0 += 32) {
    { // A tile 64x32 bf16: 256 lanes x 16B async DMA
      int r = tid >> 2, c = (tid & 3) * 8;
      unsigned voff = (unsigned)(((bm0 + r) * K + k0 + c) * 2);
      async_copy16(A, voff, &As[r][c]);
    }
    if (tid < 128) { // W tile 32x32 bf16: 128 lanes x 16B async DMA
      int r = tid >> 2, c = (tid & 3) * 8;
      unsigned voff = (unsigned)(((bn0 + r) * K + k0 + c) * 2);
      async_copy16(W, voff, &Ws[r][c]);
    }
    wait_async();
    __syncthreads();
    bf16x16 af = frag16(&As[wr * 16 + (lane & 15)][0], kbase);
    bf16x16 bf = frag16(&Ws[wc * 16 + (lane & 15)][0], kbase);
    acc = __builtin_amdgcn_wmma_f32_16x16x32_bf16(false, af, false, bf,
                                                  (short)0, acc, false, false);
    __syncthreads();
  }

  const int n = bn0 + wc * 16 + (lane & 15);
  const float bv = bias ? bias[n] : 0.f;
  const int m0 = bm0 + wr * 16 + ((lane < 16) ? 0 : 8);
  for (int r = 0; r < 8; ++r) {
    float v = acc[r] + bv;
    if (relu) v = fmaxf(v, 0.f);
    C[(size_t)(m0 + r) * Nn + n] = v;
  }
}

// ---------------------------------------------------------------------------
// Flash MHA per (graph b, head h), qkv pre-converted bf16 [N, 384] (q|k|v).
// S^T = Ktile x Q -> D-layout of S^T == B-fragment layout of P for PV WMMA.
// Block = 256 thr = 8 waves; wave owns 16 queries; block covers 128 queries.
// ---------------------------------------------------------------------------
__global__ __launch_bounds__(256) void flash_mha(
    const __bf16* __restrict__ qkv, float* __restrict__ O)
{
  __shared__ __align__(16) __bf16 Qs[128][32];
  __shared__ __align__(16) __bf16 Ks[32][32];
  __shared__ __align__(16) __bf16 Vt[32][32];   // transposed: Vt[c][key]
  const int tid = threadIdx.x, lane = tid & 31, wave = tid >> 5;
  const int b = blockIdx.x >> 2, h = blockIdx.x & 3;
  const int qblk = blockIdx.y * 128;
  const float qs = 0.17677669529663687f;  // 1/sqrt(32), applied to logits

  // stage Q 128x32 via async DMA (2 x 16B per lane)
  for (int rr = 0; rr < 2; ++rr) {
    int qr = (tid >> 2) + rr * 64, c = (tid & 3) * 8;
    unsigned voff = (unsigned)((((b * SEQ + qblk + qr) * 384) + h * 32 + c) * 2);
    async_copy16(qkv, voff, &Qs[qr][c]);
  }
  wait_async();
  __syncthreads();

  const int kbase = (lane < 16) ? 0 : 8;
  const int nq = lane & 15;
  bf16x16 qf = frag16(&Qs[wave * 16 + nq][0], kbase);

  f32x8 o0 = {}, o1 = {};
  float mrun = -INFINITY, lsum = 0.f;

  for (int kt = 0; kt < SEQ; kt += 32) {
    if (tid < 128) { // K tile 32x32 row-major via async DMA
      int r = tid >> 2, c = (tid & 3) * 8;
      unsigned voff = (unsigned)((((b * SEQ + kt + r) * 384) + HID + h * 32 + c) * 2);
      async_copy16(qkv, voff, &Ks[r][c]);
    }
    { // V tile transposed into LDS (scatter b16 stores)
      int r = tid >> 3, c = (tid & 7) * 4;
      bf16x4 vv = *(const bf16x4*)(qkv + (size_t)(b * SEQ + kt + r) * 384 + 2 * HID + h * 32 + c);
      Vt[c + 0][r] = vv[0]; Vt[c + 1][r] = vv[1];
      Vt[c + 2][r] = vv[2]; Vt[c + 3][r] = vv[3];
    }
    wait_async();
    __syncthreads();

    bf16x16 k0f = frag16(&Ks[nq][0], kbase);
    bf16x16 k1f = frag16(&Ks[16 + nq][0], kbase);
    f32x8 z = {};
    f32x8 s0 = __builtin_amdgcn_wmma_f32_16x16x32_bf16(false, k0f, false, qf, (short)0, z, false, false);
    f32x8 s1 = __builtin_amdgcn_wmma_f32_16x16x32_bf16(false, k1f, false, qf, (short)0, z, false, false);
    for (int r = 0; r < 8; ++r) { s0[r] *= qs; s1[r] *= qs; }

    // online softmax over the 32-key tile (per query column)
    float mloc = s0[0];
    for (int r = 1; r < 8; ++r) mloc = fmaxf(mloc, s0[r]);
    for (int r = 0; r < 8; ++r) mloc = fmaxf(mloc, s1[r]);
    float mtile = fmaxf(mloc, __shfl_xor(mloc, 16, 32));
    float mnew  = fmaxf(mrun, mtile);
    float scale = __expf(mrun - mnew);

    float e0[8], e1[8]; float ladd = 0.f;
    for (int r = 0; r < 8; ++r) { e0[r] = __expf(s0[r] - mnew); ladd += e0[r]; }
    for (int r = 0; r < 8; ++r) { e1[r] = __expf(s1[r] - mnew); ladd += e1[r]; }
    bf16x16 pf;
    for (int r = 0; r < 8; r += 2) {
      bf16x2 pa = pk2(e0[r], e0[r + 1]); pf[r]     = pa[0]; pf[r + 1] = pa[1];
      bf16x2 pb = pk2(e1[r], e1[r + 1]); pf[8 + r] = pb[0]; pf[9 + r] = pb[1];
    }
    ladd += __shfl_xor(ladd, 16, 32);
    lsum = lsum * scale + ladd;
    mrun = mnew;
    for (int r = 0; r < 8; ++r) { o0[r] *= scale; o1[r] *= scale; }

    bf16x16 v0f = frag16(&Vt[nq][0], kbase);        // O^T rows c=0..15
    bf16x16 v1f = frag16(&Vt[16 + nq][0], kbase);   // O^T rows c=16..31
    o0 = __builtin_amdgcn_wmma_f32_16x16x32_bf16(false, v0f, false, pf, (short)0, o0, false, false);
    o1 = __builtin_amdgcn_wmma_f32_16x16x32_bf16(false, v1f, false, pf, (short)0, o1, false, false);
    __syncthreads();
  }

  const float inv = 1.f / fmaxf(lsum, 1e-16f);
  const int node = b * SEQ + qblk + wave * 16 + nq;
  const int c0 = (lane < 16) ? 0 : 8;
  for (int r = 0; r < 8; ++r) {
    O[(size_t)node * HID + h * 32 + c0 + r]      = o0[r] * inv;
    O[(size_t)node * HID + h * 32 + 16 + c0 + r] = o1[r] * inv;
  }
}

// ---------------------------------------------------------------------------
// Edge-softmax (TransformerConv) pieces: thread per (edge, head).
// ---------------------------------------------------------------------------
__global__ void fill_f32(float* p, float v, int n) {
  for (int i = blockIdx.x * blockDim.x + threadIdx.x; i < n; i += gridDim.x * blockDim.x) p[i] = v;
}
__global__ void fill_i32(int* p, int v, int n) {
  for (int i = blockIdx.x * blockDim.x + threadIdx.x; i < n; i += gridDim.x * blockDim.x) p[i] = v;
}

__global__ void edge_logit_max(const float* __restrict__ q, const float* __restrict__ k,
                               const int* __restrict__ src, const int* __restrict__ tgt,
                               float* __restrict__ logit, int* __restrict__ mxi, int EH)
{
  int tid = blockIdx.x * blockDim.x + threadIdx.x;
  if (tid >= EH) return;
  int e = tid >> 2, h = tid & 3;
  int s = src[e], t = tgt[e];
  const f32x4* qp = (const f32x4*)(q + (size_t)t * HID + h * 32);
  const f32x4* kp = (const f32x4*)(k + (size_t)s * HID + h * 32);
  float d = 0.f;
  for (int j = 0; j < 8; ++j) {
    f32x4 a = qp[j], bb = kp[j];
    d += a[0]*bb[0] + a[1]*bb[1] + a[2]*bb[2] + a[3]*bb[3];
  }
  d *= 0.17677669529663687f;
  logit[tid] = d;
  int iv = __float_as_int(d); iv = (iv >= 0) ? iv : (iv ^ 0x7FFFFFFF);  // order-preserving map
  atomicMax(&mxi[t * 4 + h], iv);
}

__global__ void edge_exp_den(const float* __restrict__ logit, const int* __restrict__ tgt,
                             const int* __restrict__ mxi, float* __restrict__ ex,
                             float* __restrict__ den, int EH)
{
  int tid = blockIdx.x * blockDim.x + threadIdx.x;
  if (tid >= EH) return;
  int e = tid >> 2, h = tid & 3;
  int t = tgt[e];
  int iv = mxi[t * 4 + h];
  float mx = __int_as_float((iv >= 0) ? iv : (iv ^ 0x7FFFFFFF));
  float eo = __expf(logit[tid] - mx);
  ex[tid] = eo;
  atomicAdd(&den[t * 4 + h], eo);
}

__global__ void edge_agg(const float* __restrict__ ex, const float* __restrict__ den,
                         const float* __restrict__ v, const int* __restrict__ src,
                         const int* __restrict__ tgt, float* __restrict__ agg, int EH)
{
  int tid = blockIdx.x * blockDim.x + threadIdx.x;
  if (tid >= EH) return;
  int e = tid >> 2, h = tid & 3;
  int s = src[e], t = tgt[e];
  float alpha = ex[tid] / fmaxf(den[t * 4 + h], 1e-16f);
  const float* vp = v + (size_t)s * HID + h * 32;
  float* ap = agg + (size_t)t * HID + h * 32;
  for (int c = 0; c < 32; ++c) atomicAdd(&ap[c], alpha * vp[c]);
}

// beta gate (wave per node): agg <- g*r + (1-g)*agg
__global__ __launch_bounds__(256) void beta_gate(float* __restrict__ agg,
    const float* __restrict__ skip, const float* __restrict__ wbeta, int rows)
{
  int wv = (blockIdx.x * blockDim.x + threadIdx.x) >> 5;
  int lane = threadIdx.x & 31;
  if (wv >= rows) return;
  float a4[4], r4[4], d = 0.f;
  for (int j = 0; j < 4; ++j) {
    int c = lane * 4 + j;
    float a = agg[(size_t)wv * HID + c], r = skip[(size_t)wv * HID + c];
    a4[j] = a; r4[j] = r;
    d += wbeta[c] * a + wbeta[HID + c] * r + wbeta[2 * HID + c] * (a - r);
  }
  for (int off = 16; off; off >>= 1) d += __shfl_xor(d, off, 32);
  float g = 1.f / (1.f + __expf(-d));
  for (int j = 0; j < 4; ++j)
    agg[(size_t)wv * HID + lane * 4 + j] = g * r4[j] + (1.f - g) * a4[j];
}

// y = LayerNorm(a+b)*g + beta   (wave per row of 128)
__global__ __launch_bounds__(256) void ln_residual(const float* __restrict__ a,
    const float* __restrict__ b, const float* __restrict__ g, const float* __restrict__ be,
    float* __restrict__ y, int rows)
{
  int wv = (blockIdx.x * blockDim.x + threadIdx.x) >> 5;
  int lane = threadIdx.x & 31;
  if (wv >= rows) return;
  float v4[4], s = 0.f, s2 = 0.f;
  for (int j = 0; j < 4; ++j) {
    int c = lane * 4 + j;
    float t = a[(size_t)wv * HID + c] + b[(size_t)wv * HID + c];
    v4[j] = t; s += t; s2 += t * t;
  }
  for (int off = 16; off; off >>= 1) { s += __shfl_xor(s, off, 32); s2 += __shfl_xor(s2, off, 32); }
  float mean = s * (1.f / HID);
  float var  = s2 * (1.f / HID) - mean * mean;
  float rstd = rsqrtf(var + 1e-5f);
  for (int j = 0; j < 4; ++j) {
    int c = lane * 4 + j;
    y[(size_t)wv * HID + c] = (v4[j] - mean) * rstd * g[c] + be[c];
  }
}

__global__ void add_vec(const float* __restrict__ a, const float* __restrict__ b,
                        float* __restrict__ o, int n)
{
  int i = blockIdx.x * blockDim.x + threadIdx.x;
  if (i < n) o[i] = a[i] + b[i];
}

// ---------------------------------------------------------------------------
extern "C" void kernel_launch(void* const* d_in, const int* in_sizes, int n_in,
                              void* d_out, int out_size, void* d_ws, size_t ws_size,
                              hipStream_t stream)
{
  const float* x_in      = (const float*)d_in[0];
  const int*   ei        = (const int*)  d_in[1];
  const float* lin_in_w  = (const float*)d_in[3];
  const float* lin_in_b  = (const float*)d_in[4];
  const float* conv_wq   = (const float*)d_in[5];
  const float* conv_bq   = (const float*)d_in[6];
  const float* conv_wk   = (const float*)d_in[7];
  const float* conv_bk   = (const float*)d_in[8];
  const float* conv_wv   = (const float*)d_in[9];
  const float* conv_bv   = (const float*)d_in[10];
  const float* conv_wsk  = (const float*)d_in[11];
  const float* conv_bsk  = (const float*)d_in[12];
  const float* conv_wbe  = (const float*)d_in[13];
  const float* attn_in_w = (const float*)d_in[14];
  const float* attn_in_b = (const float*)d_in[15];
  const float* attn_out_w= (const float*)d_in[16];
  const float* attn_out_b= (const float*)d_in[17];
  const float* n1_g = (const float*)d_in[18]; const float* n1_b = (const float*)d_in[19];
  const float* n2_g = (const float*)d_in[20]; const float* n2_b = (const float*)d_in[21];
  const float* n3_g = (const float*)d_in[22]; const float* n3_b = (const float*)d_in[23];
  const float* mlp_w1 = (const float*)d_in[24]; const float* mlp_b1 = (const float*)d_in[25];
  const float* mlp_w2 = (const float*)d_in[26]; const float* mlp_b2 = (const float*)d_in[27];
  const float* lin_out_w = (const float*)d_in[28];
  const float* lin_out_b = (const float*)d_in[29];

  const int* src = ei;
  const int* tgt = ei + N_EDGESC;

  const size_t NM = (size_t)N_NODES * HID;     // 2,097,152 floats
  const size_t E4 = (size_t)N_EDGESC * HEADS;  // 1,048,576
  float* ws    = (float*)d_ws;
  float* xcur  = ws;
  float* xl    = ws + 1 * NM;
  float* xg    = ws + 2 * NM;
  float* qb    = ws + 3 * NM;   // q | k | v contiguous -> doubles as MHA qkv [N,384] f32
  float* kb    = ws + 4 * NM;
  float* vb    = ws + 5 * NM;
  float* qkvb  = qb;
  float* skipb = ws + 6 * NM;
  float* aggb  = ws + 7 * NM;
  float* logb  = ws + 8 * NM;
  float* exb   = logb + E4;
  int*   mxi   = (int*)(exb + E4);
  float* denb  = (float*)(mxi + N_NODES * HEADS);

  // bf16 arena (16B aligned: 19,005,440 floats precede it)
  __bf16* arena = (__bf16*)(denb + N_NODES * HEADS);
  __bf16* lin_in_wb  = arena;               arena += 128 * 128;
  __bf16* conv_wq_b  = arena;               arena += 2 * 128 * 128;
  __bf16* conv_wk_b  = arena;               arena += 2 * 128 * 128;
  __bf16* conv_wv_b  = arena;               arena += 2 * 128 * 128;
  __bf16* conv_wsk_b = arena;               arena += 2 * 128 * 128;
  __bf16* attn_in_wb = arena;               arena += 2 * 384 * 128;
  __bf16* attn_out_wb= arena;               arena += 2 * 128 * 128;
  __bf16* mlp_w1b    = arena;               arena += 2 * 128 * 128;
  __bf16* mlp_w2b    = arena;               arena += 2 * 128 * 128;
  __bf16* lin_out_wb = arena;               arena += 64 * 128;
  __bf16* xbf        = arena;               arena += NM;       // gemm A operand
  __bf16* hbf        = arena;               arena += NM;       // second activation
  __bf16* qkvbf      = arena;               arena += 3 * NM;   // flash operand

  dim3 blk(256);
  const int EH = (int)E4;

  auto cvt = [&](const float* in, __bf16* out, int n) {
    cvt_bf16<<<n / 2048, blk, 0, stream>>>(in, out, n);
  };
  auto gemm = [&](const __bf16* A, const __bf16* W, const float* bias, float* C,
                  int M, int Nn, int K, int relu) {
    dim3 grid(M / 64, Nn / 32);
    gemm_wmma<<<grid, blk, 0, stream>>>(A, W, bias, C, M, Nn, K, relu);
  };

  // weights -> bf16 once per call
  cvt(lin_in_w,   lin_in_wb,   128 * 128);
  cvt(conv_wq,    conv_wq_b,   2 * 128 * 128);
  cvt(conv_wk,    conv_wk_b,   2 * 128 * 128);
  cvt(conv_wv,    conv_wv_b,   2 * 128 * 128);
  cvt(conv_wsk,   conv_wsk_b,  2 * 128 * 128);
  cvt(attn_in_w,  attn_in_wb,  2 * 384 * 128);
  cvt(attn_out_w, attn_out_wb, 2 * 128 * 128);
  cvt(mlp_w1,     mlp_w1b,     2 * 128 * 128);
  cvt(mlp_w2,     mlp_w2b,     2 * 128 * 128);
  cvt(lin_out_w,  lin_out_wb,  64 * 128);

  // input projection
  cvt(x_in, xbf, (int)NM);
  gemm(xbf, lin_in_wb, lin_in_b, xcur, N_NODES, HID, HID, 0);

  for (int i = 0; i < 2; ++i) {
    const size_t wo = (size_t)i * HID * HID, bo = (size_t)i * HID;
    cvt(xcur, xbf, (int)NM);

    // ---- local branch: TransformerConv ----
    gemm(xbf, conv_wq_b + wo, conv_bq + bo, qb,    N_NODES, HID, HID, 0);
    gemm(xbf, conv_wk_b + wo, conv_bk + bo, kb,    N_NODES, HID, HID, 0);
    gemm(xbf, conv_wv_b + wo, conv_bv + bo, vb,    N_NODES, HID, HID, 0);
    gemm(xbf, conv_wsk_b+ wo, conv_bsk+ bo, skipb, N_NODES, HID, HID, 0);

    fill_i32<<<256, blk, 0, stream>>>(mxi, (int)0x80000000, N_NODES * HEADS);
    fill_f32<<<256, blk, 0, stream>>>(denb, 0.f, N_NODES * HEADS);
    fill_f32<<<2048, blk, 0, stream>>>(aggb, 0.f, (int)NM);

    edge_logit_max<<<EH / 256, blk, 0, stream>>>(qb, kb, src, tgt, logb, mxi, EH);
    edge_exp_den  <<<EH / 256, blk, 0, stream>>>(logb, tgt, mxi, exb, denb, EH);
    edge_agg      <<<EH / 256, blk, 0, stream>>>(exb, denb, vb, src, tgt, aggb, EH);
    beta_gate     <<<N_NODES / 8, blk, 0, stream>>>(aggb, skipb, conv_wbe + (size_t)i * 384, N_NODES);
    ln_residual   <<<N_NODES / 8, blk, 0, stream>>>(aggb, xcur, n1_g + bo, n1_b + bo, xl, N_NODES);

    // ---- global branch: dense per-graph MHA ----
    gemm(xbf, attn_in_wb + (size_t)i * 384 * HID, attn_in_b + (size_t)i * 384,
         qkvb, N_NODES, 384, HID, 0);
    cvt(qkvb, qkvbf, (int)(3 * NM));
    flash_mha<<<dim3(N_GRAPHS * HEADS, SEQ / 128), blk, 0, stream>>>(qkvbf, skipb);
    cvt(skipb, hbf, (int)NM);
    gemm(hbf, attn_out_wb + wo, attn_out_b + bo, aggb, N_NODES, HID, HID, 0);
    ln_residual<<<N_NODES / 8, blk, 0, stream>>>(aggb, xcur, n2_g + bo, n2_b + bo, xg, N_NODES);

    // ---- combine + MLP ----
    add_vec<<<(int)(NM / 256), blk, 0, stream>>>(xl, xg, xcur, (int)NM);
    cvt(xcur, xbf, (int)NM);
    gemm(xbf, mlp_w1b + wo, mlp_b1 + bo, qb, N_NODES, HID, HID, 1);
    cvt(qb, hbf, (int)NM);
    gemm(hbf, mlp_w2b + wo, mlp_b2 + bo, kb, N_NODES, HID, HID, 0);
    ln_residual<<<N_NODES / 8, blk, 0, stream>>>(xcur, kb, n3_g + bo, n3_b + bo, xcur, N_NODES);
  }

  // output projection -> d_out [16384, 64]
  cvt(xcur, xbf, (int)NM);
  gemm(xbf, lin_out_wb, lin_out_b, (float*)d_out, N_NODES, OUT_CH, HID, 0);
}
